// CrossAttention_43078521979265
// MI455X (gfx1250) — matrix operand order repaired and checked by reading, not compiled, not run on hardware
//
#include <hip/hip_runtime.h>
#include <hip/hip_bf16.h>
#include <math.h>

#define DIM 256
#define NEGV -1e9f

typedef _Float16 h16 __attribute__((ext_vector_type(16)));
typedef _Float16 h8  __attribute__((ext_vector_type(8)));
typedef float    f8  __attribute__((ext_vector_type(8)));

// ---------------------------------------------------------------------------
// Fragment loaders (wave32), contiguous k order shared by A and B (not the ISA table order).
// A-matrix 16x32 f16: lane m = lane&15; halves 0..15 -> k = koff..koff+15,
// where koff = k0 + 16*(lane>=16).
// ---------------------------------------------------------------------------
__device__ __forceinline__ h16 load_a_frag(const _Float16* base, int ldm,
                                           int lane, int k0) {
  int m    = lane & 15;
  int koff = k0 + ((lane >> 4) << 4);
  const _Float16* p = base + m * ldm + koff;
  h8 lo = *(const h8*)(p);
  h8 hi = *(const h8*)(p + 8);
  h16 a;
#pragma unroll
  for (int i = 0; i < 8; ++i) { a[i] = lo[i]; a[i + 8] = hi[i]; }
  return a;
}

// B-matrix 32x16 f16: lane col n = lane&15, 16 contiguous k-halves starting at
// k = k0 + 16*(lane>=16). Caller passes fully-resolved pointer.
__device__ __forceinline__ h16 load_b_frag(const _Float16* p) {
  return *(const h16*)p;
}

// ---------------------------------------------------------------------------
// fp32 -> f16 conversion (weights, one pass). 4 elements / thread.
// ---------------------------------------------------------------------------
__global__ void __launch_bounds__(256)
cvt_f16(const float* __restrict__ src, _Float16* __restrict__ dst) {
  int i = (blockIdx.x * 256 + threadIdx.x) * 4;
  float4 v = *(const float4*)(src + i);
  dst[i + 0] = (_Float16)v.x;
  dst[i + 1] = (_Float16)v.y;
  dst[i + 2] = (_Float16)v.z;
  dst[i + 3] = (_Float16)v.w;
}

// ---------------------------------------------------------------------------
// GEMM: Y[m][n] = sum_k X[m][k] * Wh[n][k]  (X @ W.T), X fp32, Wh f16.
// Block = 128 threads (4 waves), 64 rows / block; blockIdx.y splits the 16
// n-tiles in two (8 per block) to double grid occupancy. B fragments are
// preloaded as a register array so the 16 b128 loads clause and overlap the
// dependent WMMA chain.
// ---------------------------------------------------------------------------
template <bool HALF_OUT>
__global__ void __launch_bounds__(128)
gemm16(const float* __restrict__ X, const _Float16* __restrict__ Wh,
       void* __restrict__ Yout) {
  __shared__ alignas(16) _Float16 Xlds[64 * DIM];
  const int tid = threadIdx.x, wave = tid >> 5, lane = tid & 31;
  const int rowBase = blockIdx.x * 64;

  // Stage 64x256 fp32 -> f16 into LDS.
  for (int i = tid; i < 64 * DIM / 4; i += 128) {
    float4 v = ((const float4*)(X + rowBase * DIM))[i];
    int o = i * 4;
    Xlds[o + 0] = (_Float16)v.x; Xlds[o + 1] = (_Float16)v.y;
    Xlds[o + 2] = (_Float16)v.z; Xlds[o + 3] = (_Float16)v.w;
  }
  __syncthreads();

  const _Float16* xw = Xlds + (wave * 16) * DIM;
  h16 A[8];
#pragma unroll
  for (int kc = 0; kc < 8; ++kc) A[kc] = load_a_frag(xw, DIM, lane, kc * 32);

  const int n = lane & 15, hg = lane >> 4;
  const int nbBegin = blockIdx.y * 8;
  for (int nb = nbBegin; nb < nbBegin + 8; ++nb) {
    h16 B[8];
#pragma unroll
    for (int kc = 0; kc < 8; ++kc)
      B[kc] = load_b_frag(Wh + (nb * 16 + n) * DIM + kc * 32 + 16 * hg);
    f8 acc = {};
#pragma unroll
    for (int kc = 0; kc < 8; ++kc)
      acc = __builtin_amdgcn_wmma_f32_16x16x32_f16(false, A[kc], false, B[kc],
                                                   (short)0, acc, false, false);
#pragma unroll
    for (int r = 0; r < 8; ++r) {
      int row = rowBase + wave * 16 + r + 8 * hg;
      int col = nb * 16 + n;
      if constexpr (HALF_OUT)
        ((_Float16*)Yout)[row * DIM + col] = (_Float16)acc[r];
      else
        ((float*)Yout)[row * DIM + col] = acc[r];
    }
  }
}

// ---------------------------------------------------------------------------
// Flash cross-attention with split-K. Block = 128 threads (4 waves) = 64
// queries; blockIdx.y picks a key range of kPerSplit keys. Each split writes
// unnormalized O plus per-row (m, l); a combine kernel merges the splits.
// K chunk staged with global_load_async_to_lds_b128 (ASYNCcnt path).
// ---------------------------------------------------------------------------
__global__ void __launch_bounds__(128)
attn(const _Float16* __restrict__ Qh, const _Float16* __restrict__ Kh,
     const _Float16* __restrict__ Vh, const float* __restrict__ qpos,
     const float* __restrict__ kpos, const int* __restrict__ qbatch,
     const int* __restrict__ kbatch, int kPerSplit, int NQ,
     float* __restrict__ Opart, float* __restrict__ mpart,
     float* __restrict__ lpart) {
  __shared__ alignas(16) _Float16 Klds[32 * DIM];   // 16 KB: K chunk row-major
  __shared__ alignas(16) _Float16 Vt[DIM * 32];     // 16 KB: V chunk transposed
  __shared__ alignas(16) _Float16 Plds[4][16 * 32]; // 4 KB: P re-layout buffer
  __shared__ float qpl[64 * 3];
  __shared__ int   qbl[64];
  __shared__ float kpl[32 * 3];
  __shared__ int   kbl[32];

  const int tid = threadIdx.x, wave = tid >> 5, lane = tid & 31;
  const int qBase = blockIdx.x * 64;
  const int split = blockIdx.y;
  const int kBegin = split * kPerSplit, kEnd = kBegin + kPerSplit;
  const int n = lane & 15, hg = lane >> 4;

  for (int i = tid; i < 64; i += 128) {
    qbl[i] = qbatch[qBase + i];
    qpl[i * 3 + 0] = qpos[(qBase + i) * 3 + 0];
    qpl[i * 3 + 1] = qpos[(qBase + i) * 3 + 1];
    qpl[i * 3 + 2] = qpos[(qBase + i) * 3 + 2];
  }

  // Resident Q fragments (16 rows x 256 k, 8 A-fragments).
  h16 A[8];
  const _Float16* qrow = Qh + (qBase + wave * 16) * DIM;
#pragma unroll
  for (int kc = 0; kc < 8; ++kc) A[kc] = load_a_frag(qrow, DIM, lane, kc * 32);

  __syncthreads();
  // Hoist per-row query metadata into registers (rows r -> m = r + 8*hg).
  float qxr[8], qyr[8], qzr[8];
  int qbr[8];
#pragma unroll
  for (int r = 0; r < 8; ++r) {
    int lm = wave * 16 + r + 8 * hg;
    qxr[r] = qpl[lm * 3 + 0];
    qyr[r] = qpl[lm * 3 + 1];
    qzr[r] = qpl[lm * 3 + 2];
    qbr[r] = qbl[lm];
  }

  f8 O[16];
#pragma unroll
  for (int t = 0; t < 16; ++t) O[t] = (f8){};
  float mrow[8], lrow[8];
#pragma unroll
  for (int r = 0; r < 8; ++r) { mrow[r] = -3.0e38f; lrow[r] = 0.0f; }

  const unsigned kldsBase = (unsigned)(uintptr_t)(void*)Klds;

  for (int k0 = kBegin; k0 < kEnd; k0 += 32) {
    __syncthreads();

    // --- K chunk: async copy global -> LDS (16B per issue, 8 per thread) ---
    {
      unsigned long long gbase =
          (unsigned long long)(uintptr_t)(const void*)(Kh + (size_t)k0 * DIM);
      for (int i = tid; i < 32 * DIM / 8; i += 128) {
        unsigned la = kldsBase + (unsigned)(i * 16);
        unsigned long long ga = gbase + (unsigned long long)i * 16;
        asm volatile("global_load_async_to_lds_b128 %0, %1, off"
                     :: "v"(la), "v"(ga) : "memory");
      }
    }

    // --- V chunk transposed: vectorized global loads, scattered LDS stores ---
    for (int i = tid; i < 32 * (DIM / 8); i += 128) {
      int kk = i >> 5;         // V row within chunk
      int j  = i & 31;         // uint4 index within row
      uint4 raw = ((const uint4*)(Vh + (size_t)(k0 + kk) * DIM))[j];
      _Float16 tmp[8];
      *(uint4*)tmp = raw;
#pragma unroll
      for (int e = 0; e < 8; ++e) Vt[(j * 8 + e) * 32 + kk] = tmp[e];
    }

    // --- key metadata ---
    for (int i = tid; i < 32; i += 128) {
      kbl[i] = kbatch[k0 + i];
      kpl[i * 3 + 0] = kpos[(k0 + i) * 3 + 0];
      kpl[i * 3 + 1] = kpos[(k0 + i) * 3 + 1];
      kpl[i * 3 + 2] = kpos[(k0 + i) * 3 + 2];
    }
    if (k0 + 32 < kEnd)  // prefetch next K chunk (global_prefetch_b8)
      __builtin_prefetch(Kh + (size_t)(k0 + 32) * DIM, 0, 1);

    asm volatile("s_wait_asynccnt 0x0" ::: "memory");
    __syncthreads();

    // Scores: S[2] tiles of 16x16 over 32 keys.
    f8 S[2];
#pragma unroll
    for (int nb = 0; nb < 2; ++nb) {
      f8 acc = {};
#pragma unroll
      for (int kc = 0; kc < 8; ++kc) {
        const _Float16* bp = Klds + (nb * 16 + n) * DIM + kc * 32 + 16 * hg;
        acc = __builtin_amdgcn_wmma_f32_16x16x32_f16(
            false, A[kc], false, load_b_frag(bp), (short)0, acc, false, false);
      }
      S[nb] = acc;
    }

    // Per-column key metadata for the two score tiles.
    const float kx0 = kpl[n * 3 + 0], ky0 = kpl[n * 3 + 1], kz0 = kpl[n * 3 + 2];
    const float kx1 = kpl[(16 + n) * 3 + 0], ky1 = kpl[(16 + n) * 3 + 1],
                kz1 = kpl[(16 + n) * 3 + 2];
    const int kb0 = kbl[n], kb1 = kbl[16 + n];

#pragma unroll
    for (int r = 0; r < 8; ++r) {
      const float qx = qxr[r], qy = qyr[r], qz = qzr[r];
      const int qb = qbr[r];

      float d2, v0, v1;
      d2 = (qx - kx0) * (qx - kx0) + (qy - ky0) * (qy - ky0) +
           (qz - kz0) * (qz - kz0);
      v0 = (qb == kb0)
               ? S[0][r] * 0.0625f + __expf(-sqrtf(fmaxf(d2, 1e-12f)) * 0.1f)
               : NEGV;
      d2 = (qx - kx1) * (qx - kx1) + (qy - ky1) * (qy - ky1) +
           (qz - kz1) * (qz - kz1);
      v1 = (qb == kb1)
               ? S[1][r] * 0.0625f + __expf(-sqrtf(fmaxf(d2, 1e-12f)) * 0.1f)
               : NEGV;

      // Online softmax: reductions across the 16-lane column group.
      float mx = fmaxf(v0, v1);
#pragma unroll
      for (int s = 8; s; s >>= 1) mx = fmaxf(mx, __shfl_xor(mx, s, 32));
      const float mN = fmaxf(mrow[r], mx);
      const float alpha = __expf(mrow[r] - mN);
      const float p0 = __expf(v0 - mN), p1 = __expf(v1 - mN);
      float ps = p0 + p1;
#pragma unroll
      for (int s = 8; s; s >>= 1) ps += __shfl_xor(ps, s, 32);
      lrow[r] = lrow[r] * alpha + ps;
      mrow[r] = mN;
#pragma unroll
      for (int t = 0; t < 16; ++t) O[t][r] *= alpha;

      Plds[wave][(r + 8 * hg) * 32 + n]      = (_Float16)p0;
      Plds[wave][(r + 8 * hg) * 32 + 16 + n] = (_Float16)p1;
    }

    // Wave-local LDS RAW: P written in C-layout, read back in A-layout.
    asm volatile("s_wait_dscnt 0x0" ::: "memory");
    h16 Pa = load_a_frag(&Plds[wave][0], 32, lane, 0);
#pragma unroll
    for (int t = 0; t < 16; ++t) {
      const _Float16* bv = Vt + (t * 16 + n) * 32 + 16 * hg;
      O[t] = __builtin_amdgcn_wmma_f32_16x16x32_f16(
          false, Pa, false, load_b_frag(bv), (short)0, O[t], false, false);
    }
  }

  // Write unnormalized partial O and per-row (m, l) for this split.
  float* Osplit = Opart + (size_t)split * NQ * DIM;
#pragma unroll
  for (int t = 0; t < 16; ++t)
#pragma unroll
    for (int r = 0; r < 8; ++r) {
      int row = qBase + wave * 16 + r + 8 * hg;
      Osplit[(size_t)row * DIM + t * 16 + n] = O[t][r];
    }
  if (n == 0) {
#pragma unroll
    for (int r = 0; r < 8; ++r) {
      int row = qBase + wave * 16 + r + 8 * hg;
      mpart[(size_t)split * NQ + row] = mrow[r];
      lpart[(size_t)split * NQ + row] = lrow[r];
    }
  }
}

// ---------------------------------------------------------------------------
// Merge split-K partials: ctx = (sum_s e^(m_s-M) O_s) / (sum_s e^(m_s-M) l_s)
// ---------------------------------------------------------------------------
__global__ void __launch_bounds__(256)
combine(const float* __restrict__ Opart, const float* __restrict__ mpart,
        const float* __restrict__ lpart, int NQ, int S,
        float* __restrict__ ctx) {
  const int row = blockIdx.x, tid = threadIdx.x;
  float M = -3.0e38f;
  for (int s = 0; s < S; ++s) M = fmaxf(M, mpart[(size_t)s * NQ + row]);
  float acc = 0.0f, lsum = 0.0f;
  for (int s = 0; s < S; ++s) {
    float w = __expf(mpart[(size_t)s * NQ + row] - M);
    lsum += w * lpart[(size_t)s * NQ + row];
    acc += w * Opart[((size_t)s * NQ + row) * DIM + tid];
  }
  ctx[(size_t)row * DIM + tid] = acc / lsum;
}

// ---------------------------------------------------------------------------
// out = layernorm(x + proj + bias) * g + b  per row of 256.
// ---------------------------------------------------------------------------
__global__ void __launch_bounds__(256)
ln_kernel(const float* __restrict__ x, const float* __restrict__ proj,
          const float* __restrict__ bias, const float* __restrict__ g,
          const float* __restrict__ b, float* __restrict__ out) {
  __shared__ float red[256];
  const int row = blockIdx.x, tid = threadIdx.x;
  const float v = x[row * DIM + tid] + proj[row * DIM + tid] + bias[tid];
  red[tid] = v;
  __syncthreads();
  for (int s = 128; s; s >>= 1) {
    if (tid < s) red[tid] += red[tid + s];
    __syncthreads();
  }
  const float mu = red[0] * (1.0f / DIM);
  __syncthreads();
  const float d = v - mu;
  red[tid] = d * d;
  __syncthreads();
  for (int s = 128; s; s >>= 1) {
    if (tid < s) red[tid] += red[tid + s];
    __syncthreads();
  }
  const float var = red[0] * (1.0f / DIM);
  out[row * DIM + tid] = d * rsqrtf(var + 1e-5f) * g[tid] + b[tid];
}

// ---------------------------------------------------------------------------
extern "C" void kernel_launch(void* const* d_in, const int* in_sizes, int n_in,
                              void* d_out, int out_size, void* d_ws,
                              size_t ws_size, hipStream_t stream) {
  const int NL = 2048, NP = 16384;
  const int SPL = 8;  // key splits for lig<-pro (NK = 16384)
  const int SPP = 2;  // key splits for pro<-lig (NK = 2048)
  const float* lig      = (const float*)d_in[0];
  const float* pro      = (const float*)d_in[1];
  const float* lig_pos  = (const float*)d_in[2];
  const float* pro_pos  = (const float*)d_in[3];
  const int*   lig_bat  = (const int*)d_in[4];
  const int*   pro_bat  = (const int*)d_in[5];
  const float* Wmats[8] = {(const float*)d_in[6],  (const float*)d_in[7],
                           (const float*)d_in[8],  (const float*)d_in[9],
                           (const float*)d_in[10], (const float*)d_in[11],
                           (const float*)d_in[12], (const float*)d_in[14]};
  const float* bout_lig = (const float*)d_in[13];
  const float* bout_pro = (const float*)d_in[15];
  const float* g_lig    = (const float*)d_in[16];
  const float* b_lig    = (const float*)d_in[17];
  const float* g_pro    = (const float*)d_in[18];
  const float* b_pro    = (const float*)d_in[19];
  float* out = (float*)d_out;

  // Workspace carve-up.
  char* ws = (char*)d_ws;
  size_t off = 0;
  auto take = [&](size_t bytes) {
    char* p = ws + off;
    off += (bytes + 255) & ~size_t(255);
    return p;
  };
  _Float16* Wh[8];
  for (int i = 0; i < 8; ++i) Wh[i] = (_Float16*)take((size_t)DIM * DIM * 2);
  _Float16* Qlig = (_Float16*)take((size_t)NL * DIM * 2);
  _Float16* Klig = (_Float16*)take((size_t)NL * DIM * 2);
  _Float16* Vlig = (_Float16*)take((size_t)NL * DIM * 2);
  _Float16* Kpro = (_Float16*)take((size_t)NP * DIM * 2);
  _Float16* Vpro = (_Float16*)take((size_t)NP * DIM * 2);
  _Float16* Qpro = (_Float16*)take((size_t)NP * DIM * 2);
  float* lctx  = (float*)take((size_t)NL * DIM * 4);
  float* pctx  = (float*)take((size_t)NP * DIM * 4);
  float* lproj = (float*)take((size_t)NL * DIM * 4);
  float* pproj = (float*)take((size_t)NP * DIM * 4);
  float* OpL = (float*)take((size_t)SPL * NL * DIM * 4);
  float* mL  = (float*)take((size_t)SPL * NL * 4);
  float* lL  = (float*)take((size_t)SPL * NL * 4);
  float* OpP = (float*)take((size_t)SPP * NP * DIM * 4);
  float* mP  = (float*)take((size_t)SPP * NP * 4);
  float* lP  = (float*)take((size_t)SPP * NP * 4);

  // Pre-convert weight matrices to f16 (one pass each).
  const int cvtGrid = DIM * DIM / 1024;
  for (int i = 0; i < 8; ++i)
    cvt_f16<<<cvtGrid, 256, 0, stream>>>(Wmats[i], Wh[i]);

  const dim3 blk(128);
  // Input projections (X @ W.T), f16 outputs for WMMA attention.
  gemm16<true><<<dim3(NL / 64, 2), blk, 0, stream>>>(lig, Wh[0], Qlig);
  gemm16<true><<<dim3(NP / 64, 2), blk, 0, stream>>>(pro, Wh[1], Kpro);
  gemm16<true><<<dim3(NP / 64, 2), blk, 0, stream>>>(pro, Wh[2], Vpro);
  gemm16<true><<<dim3(NP / 64, 2), blk, 0, stream>>>(pro, Wh[3], Qpro);
  gemm16<true><<<dim3(NL / 64, 2), blk, 0, stream>>>(lig, Wh[4], Klig);
  gemm16<true><<<dim3(NL / 64, 2), blk, 0, stream>>>(lig, Wh[5], Vlig);

  // Flash cross-attention with split-K, both directions.
  attn<<<dim3(NL / 64, SPL), blk, 0, stream>>>(Qlig, Kpro, Vpro, lig_pos,
                                               pro_pos, lig_bat, pro_bat,
                                               NP / SPL, NL, OpL, mL, lL);
  attn<<<dim3(NP / 64, SPP), blk, 0, stream>>>(Qpro, Klig, Vlig, pro_pos,
                                               lig_pos, pro_bat, lig_bat,
                                               NL / SPP, NP, OpP, mP, lP);
  combine<<<NL, 256, 0, stream>>>(OpL, mL, lL, NL, SPL, lctx);
  combine<<<NP, 256, 0, stream>>>(OpP, mP, lP, NP, SPP, pctx);

  // Output projections (ctx @ Wout.T), fp32 outputs.
  gemm16<false><<<dim3(NL / 64, 2), blk, 0, stream>>>(lctx, Wh[6], lproj);
  gemm16<false><<<dim3(NP / 64, 2), blk, 0, stream>>>(pctx, Wh[7], pproj);

  // Residual + bias + layernorm.
  ln_kernel<<<NL, 256, 0, stream>>>(lig, lproj, bout_lig, g_lig, b_lig, out);
  ln_kernel<<<NP, 256, 0, stream>>>(pro, pproj, bout_pro, g_pro, b_pro,
                                    out + (size_t)NL * DIM);
}
